// RG_LSTM_45217415692967
// MI455X (gfx1250) — matrix operand order repaired and checked by reading, not compiled, not run on hardware
//
#include <hip/hip_runtime.h>
#include <hip/hip_bf16.h>
#include <math.h>

// Problem dims (fixed by the reference)
#define N_NODES 2000
#define MROWS   2016   // node rows padded to 63*32 for 32-row M-blocking
#define NPAD    2048   // K padding for the A_T @ q GEMM (2000 -> 2048)
#define HDIM    256
#define TSTEPS  200
#define FDIM    16
#define FSDIM   32
#define KCAT    320    // FD + FS + H = 304, padded to 320 (10 * 32)
#define G4H     1024   // 4 * H

typedef __attribute__((ext_vector_type(16))) __bf16       v16bf;
typedef __attribute__((ext_vector_type(8)))  float        v8f;
typedef __attribute__((ext_vector_type(4)))  unsigned int u32x4;

__device__ __forceinline__ float sigmoidf_(float x) { return 1.0f / (1.0f + expf(-x)); }

// ---------------------------------------------------------------------------
// Utility zero-fill kernels (d_ws is poisoned by the harness; state must be 0)
// ---------------------------------------------------------------------------
__global__ void zero_f32(float* p, long n) {
    long i = (long)blockIdx.x * blockDim.x + threadIdx.x;
    if (i < n) p[i] = 0.0f;
}
__global__ void zero_bf16(__bf16* p, long n) {
    long i = (long)blockIdx.x * blockDim.x + threadIdx.x;
    if (i < n) p[i] = (__bf16)0.0f;
}

// ---------------------------------------------------------------------------
// NaN-aware stats over dist (zeros excluded): count / sum / sum-of-squares
// ---------------------------------------------------------------------------
__global__ void stats_kernel(const float* __restrict__ dist, long n, float* __restrict__ stats) {
    __shared__ float sc[256], ss[256], sq[256];
    long tid0   = (long)blockIdx.x * blockDim.x + threadIdx.x;
    long stride = (long)gridDim.x * blockDim.x;
    float c = 0.0f, s = 0.0f, q = 0.0f;
    for (long i = tid0; i < n; i += stride) {
        float d = dist[i];
        if (d != 0.0f) { c += 1.0f; s += d; q += d * d; }
    }
    int t = threadIdx.x;
    sc[t] = c; ss[t] = s; sq[t] = q;
    __syncthreads();
    for (int off = 128; off > 0; off >>= 1) {
        if (t < off) { sc[t] += sc[t + off]; ss[t] += ss[t + off]; sq[t] += sq[t + off]; }
        __syncthreads();
    }
    if (t == 0) {
        atomicAdd(&stats[0], sc[0]);
        atomicAdd(&stats[1], ss[0]);
        atomicAdd(&stats[2], sq[0]);
    }
}

// A_T[j][i] = logistic(-(dist[i][j]-mean)/std), zeros -> 0, stored bf16,
// [MROWS x NPAD]: rows 2000..2015 and columns 2000..2047 zero-padded.
__global__ void adj_kernel(const float* __restrict__ dist, const float* __restrict__ stats,
                           __bf16* __restrict__ At) {
    long idx = (long)blockIdx.x * blockDim.x + threadIdx.x;
    if (idx >= (long)MROWS * NPAD) return;
    int j = (int)(idx / NPAD);
    int i = (int)(idx % NPAD);
    float a = 0.0f;
    if (i < N_NODES && j < N_NODES) {
        float d = dist[(size_t)i * N_NODES + j];
        if (d != 0.0f) {
            float cnt  = stats[0];
            float mean = stats[1] / cnt;
            float var  = stats[2] / cnt - mean * mean;   // nanstd, ddof=0
            float dn   = (d - mean) / sqrtf(var);
            a = 1.0f / (1.0f + expf(dn));
        }
    }
    At[idx] = (__bf16)a;
}

// ---------------------------------------------------------------------------
// One-time weight packing (bf16, transposed for B-operand layout)
// ---------------------------------------------------------------------------
__global__ void wqT_kernel(const float* __restrict__ wq, __bf16* __restrict__ wqT) {
    int idx = blockIdx.x * blockDim.x + threadIdx.x;  // 256*256
    if (idx >= HDIM * HDIM) return;
    int k = idx / HDIM, j = idx % HDIM;
    wqT[idx] = (__bf16)wq[(size_t)j * HDIM + k];      // B[k][j] = wq[j][k]
}

// WcatT [KCAT=320, 4H=1024]: rows 0..47 = w_ih cols, rows 48..303 = w_hh cols, rest 0
__global__ void wcat_kernel(const float* __restrict__ w_ih, const float* __restrict__ w_hh,
                            __bf16* __restrict__ wcatT) {
    int idx = blockIdx.x * blockDim.x + threadIdx.x;  // 320*1024
    if (idx >= KCAT * G4H) return;
    int k = idx / G4H, g = idx % G4H;
    float v = 0.0f;
    if (k < FDIM + FSDIM)             v = w_ih[(size_t)g * (FDIM + FSDIM) + k];
    else if (k < FDIM + FSDIM + HDIM) v = w_hh[(size_t)g * HDIM + (k - FDIM - FSDIM)];
    wcatT[idx] = (__bf16)v;
}

// xcat columns 16..47 = static x_s
__global__ void pack_xs_kernel(const float* __restrict__ x_s, __bf16* __restrict__ xcat) {
    int idx = blockIdx.x * blockDim.x + threadIdx.x;  // 2000*32
    if (idx >= N_NODES * FSDIM) return;
    int n = idx / FSDIM, k = idx % FSDIM;
    xcat[(size_t)n * KCAT + FDIM + k] = (__bf16)x_s[idx];
}

// per-step: xcat columns 0..15 = x_d[t]
__global__ void pack_xd_kernel(const float* __restrict__ x_d, int t, __bf16* __restrict__ xcat) {
    int idx = blockIdx.x * blockDim.x + threadIdx.x;  // 2000*16
    if (idx >= N_NODES * FDIM) return;
    int n = idx / FDIM, k = idx % FDIM;
    xcat[(size_t)n * KCAT + k] = (__bf16)x_d[(size_t)t * N_NODES * FDIM + idx];
}

// ---------------------------------------------------------------------------
// Register-blocked bf16 WMMA GEMM: each wave owns a 32x64 output tile
// (2 M-subtiles x 4 N-subtiles = 8 fp32 accumulators). Per K-chunk (K=32):
// 2 A-tile loads (4 x b128) + 4 B-tiles (16 strided bf16 loads each) feed 8 v_wmma.
//   A-operand layout: lane (l&15) = row M; halves hold K {0..7,16..23}+8*(l>>4)
//   B-operand layout: lane (l&15) = col N; halves hold K {0..7,16..23}+8*(l>>4)
// MODE 0: q  = bf16(tanh(acc + bias[n]))      (__bf16 out)
// MODE 1: diff = acc                           (float out)
// MODE 2: gates = acc + bias[n]                (float out)
// ---------------------------------------------------------------------------
template <int MODE>
__global__ __launch_bounds__(256) void wmma_gemm_kernel(
    const __bf16* __restrict__ A, int lda,
    const __bf16* __restrict__ B, int ldb,
    int kChunks, int tilesX, int nTiles,
    const float* __restrict__ bias,
    void* __restrict__ outp, int ldo)
{
    const int wave = (blockIdx.x * blockDim.x + threadIdx.x) >> 5;
    if (wave >= nTiles) return;                 // wave-uniform: EXEC stays all-1s
    const int lane = threadIdx.x & 31;
    const int half = lane >> 4;
    const int l15  = lane & 15;
    const int m0 = (wave / tilesX) * 32;
    const int n0 = (wave % tilesX) * 64;

    const __bf16* arow = A + (size_t)(m0 + l15) * lda + half * 8;
    const __bf16* brow = B + (size_t)(half * 8) * ldb + n0 + l15;
    const size_t  astp = (size_t)16 * lda;      // second M-subtile
    const size_t  bstep = (size_t)32 * ldb;     // next K-chunk of B

    v8f acc[2][4] = {};
    for (int k = 0; k < kChunks; ++k) {
        union { v16bf v; u32x4 u[2]; } a[2], bb[4];
        a[0].u[0] = *(const u32x4*)(arow);              // K = k*32 + 8*half + 0..7
        a[0].u[1] = *(const u32x4*)(arow + 16);         // K = k*32 + 8*half + 16..23
        a[1].u[0] = *(const u32x4*)(arow + astp);
        a[1].u[1] = *(const u32x4*)(arow + astp + 16);
        #pragma unroll
        for (int j = 0; j < 4; ++j) {
            #pragma unroll
            for (int s = 0; s < 8; ++s) {
                bb[j].v[s]     = brow[(size_t)s * ldb + 16 * j];         // K = k*32 + 8*half + s
                bb[j].v[8 + s] = brow[(size_t)(16 + s) * ldb + 16 * j];  // K = k*32 + 8*half + 16 + s
            }
        }
        arow += 32;
        brow += bstep;
        if (k + 1 < kChunks) {                  // prefetch next chunk toward near cache
            __builtin_prefetch(arow, 0, 3);
            __builtin_prefetch(arow + astp, 0, 3);
            __builtin_prefetch(brow, 0, 3);
        }
        #pragma unroll
        for (int mi = 0; mi < 2; ++mi)
            #pragma unroll
            for (int ni = 0; ni < 4; ++ni)
                acc[mi][ni] = __builtin_amdgcn_wmma_f32_16x16x32_bf16(
                    false, a[mi].v, false, bb[ni].v, (short)0, acc[mi][ni], false, false);
    }

    #pragma unroll
    for (int mi = 0; mi < 2; ++mi) {
        #pragma unroll
        for (int ni = 0; ni < 4; ++ni) {
            const int n = n0 + ni * 16 + l15;            // C/D: N = lane&15
            #pragma unroll
            for (int r = 0; r < 8; ++r) {
                const int row = m0 + mi * 16 + r + 8 * half;  // C/D: M = r (+8 upper half)
                const float v = acc[mi][ni][r];
                if (MODE == 0) {
                    ((__bf16*)outp)[(size_t)row * ldo + n] = (__bf16)tanhf(v + bias[n]);
                } else if (MODE == 1) {
                    ((float*)outp)[(size_t)row * ldo + n] = v;
                } else {
                    ((float*)outp)[(size_t)row * ldo + n] = v + bias[n];
                }
            }
        }
    }
}

// ---------------------------------------------------------------------------
// Pointwise LSTM update; also refreshes the bf16 copies of h used as GEMM input
// ---------------------------------------------------------------------------
__global__ void update_kernel(const float* __restrict__ gates, const float* __restrict__ diff,
                              float* __restrict__ c, float* __restrict__ h,
                              __bf16* __restrict__ hbf, __bf16* __restrict__ xcat) {
    int idx = blockIdx.x * blockDim.x + threadIdx.x;
    if (idx >= N_NODES * HDIM) return;
    int n = idx / HDIM, j = idx % HDIM;
    const float* g = gates + (size_t)n * G4H;
    float ig = sigmoidf_(g[j]);
    float fg = sigmoidf_(g[j + 256]);
    float gg = tanhf(g[j + 512]);
    float og = sigmoidf_(g[j + 768]);
    float c0 = c[idx] + diff[idx];
    float cn = fg * c0 + ig * gg;
    float hn = og * tanhf(cn);
    c[idx] = cn;
    h[idx] = hn;
    __bf16 hb = (__bf16)hn;
    hbf[idx] = hb;
    xcat[(size_t)n * KCAT + (FDIM + FSDIM) + j] = hb;
}

// Final projection: out[n] = h[n] . wd + bd   (fp32, trivial)
__global__ void out_kernel(const float* __restrict__ h, const float* __restrict__ wd,
                           const float* __restrict__ bd, float* __restrict__ out) {
    __shared__ float s[256];
    int n = blockIdx.x, t = threadIdx.x;
    s[t] = h[(size_t)n * HDIM + t] * wd[t];
    __syncthreads();
    for (int off = 128; off > 0; off >>= 1) {
        if (t < off) s[t] += s[t + off];
        __syncthreads();
    }
    if (t == 0) out[n] = s[0] + bd[0];
}

// ---------------------------------------------------------------------------
extern "C" void kernel_launch(void* const* d_in, const int* in_sizes, int n_in,
                              void* d_out, int out_size, void* d_ws, size_t ws_size,
                              hipStream_t stream) {
    const float* dist = (const float*)d_in[0];
    const float* x_d  = (const float*)d_in[1];
    const float* x_s  = (const float*)d_in[2];
    const float* w_ih = (const float*)d_in[3];
    const float* w_hh = (const float*)d_in[4];
    const float* b    = (const float*)d_in[5];
    const float* wq   = (const float*)d_in[6];
    const float* bq   = (const float*)d_in[7];
    const float* wd   = (const float*)d_in[8];
    const float* bd   = (const float*)d_in[9];

    char* p = (char*)d_ws;
    auto alloc = [&](size_t bytes) -> void* {
        void* r = (void*)p;
        p += (bytes + 255) & ~(size_t)255;
        return r;
    };
    float*  stats = (float*) alloc(4 * sizeof(float));
    __bf16* At    = (__bf16*)alloc((size_t)MROWS * NPAD * 2);    // ~8.3 MB, L2-resident
    __bf16* q     = (__bf16*)alloc((size_t)NPAD * HDIM * 2);     // K-padded rows
    float*  h     = (float*) alloc((size_t)N_NODES * HDIM * 4);
    float*  c     = (float*) alloc((size_t)N_NODES * HDIM * 4);
    __bf16* hbf   = (__bf16*)alloc((size_t)MROWS * HDIM * 2);
    __bf16* xcat  = (__bf16*)alloc((size_t)MROWS * KCAT * 2);
    float*  gates = (float*) alloc((size_t)MROWS * G4H * 4);
    float*  diff  = (float*) alloc((size_t)MROWS * HDIM * 4);
    __bf16* wqT   = (__bf16*)alloc((size_t)HDIM * HDIM * 2);
    __bf16* wcatT = (__bf16*)alloc((size_t)KCAT * G4H * 2);

    auto gr = [](long n) { return dim3((unsigned)((n + 255) / 256)); };

    // --- one-time prep (re-run every call: kernel_launch must be deterministic)
    zero_f32 <<<gr(4), 256, 0, stream>>>(stats, 4);
    zero_f32 <<<gr((long)N_NODES * HDIM), 256, 0, stream>>>(h, (long)N_NODES * HDIM);
    zero_f32 <<<gr((long)N_NODES * HDIM), 256, 0, stream>>>(c, (long)N_NODES * HDIM);
    zero_bf16<<<gr((long)MROWS * HDIM), 256, 0, stream>>>(hbf, (long)MROWS * HDIM);
    zero_bf16<<<gr((long)MROWS * KCAT), 256, 0, stream>>>(xcat, (long)MROWS * KCAT);
    zero_bf16<<<gr((long)(NPAD - N_NODES) * HDIM), 256, 0, stream>>>(
        q + (size_t)N_NODES * HDIM, (long)(NPAD - N_NODES) * HDIM);

    stats_kernel<<<512, 256, 0, stream>>>(dist, (long)N_NODES * N_NODES, stats);
    adj_kernel  <<<gr((long)MROWS * NPAD), 256, 0, stream>>>(dist, stats, At);
    wqT_kernel  <<<gr(HDIM * HDIM), 256, 0, stream>>>(wq, wqT);
    wcat_kernel <<<gr(KCAT * G4H), 256, 0, stream>>>(w_ih, w_hh, wcatT);
    pack_xs_kernel<<<gr(N_NODES * FSDIM), 256, 0, stream>>>(x_s, xcat);

    // tile counts (32x64 tiles): rows = MROWS/32 = 63
    const int qTiles = 63 * (HDIM / 64);   // 252
    const int gTiles = 63 * (G4H / 64);    // 1008

    // --- recurrence: T sequential steps, all GEMMs on the bf16 WMMA path
    for (int t = 0; t < TSTEPS; ++t) {
        pack_xd_kernel<<<gr(N_NODES * FDIM), 256, 0, stream>>>(x_d, t, xcat);
        // q = tanh(h @ wq^T + bq)   [2016x256], K=256
        wmma_gemm_kernel<0><<<gr((long)qTiles * 32), 256, 0, stream>>>(
            hbf, HDIM, wqT, HDIM, 8, HDIM / 64, qTiles, bq, q, HDIM);
        // diff = A^T @ q            [2016x256], K=2048 (padded)
        wmma_gemm_kernel<1><<<gr((long)qTiles * 32), 256, 0, stream>>>(
            At, NPAD, q, HDIM, 64, HDIM / 64, qTiles, nullptr, diff, HDIM);
        // gates = [x_d,x_s,h] @ Wcat^T + b  [2016x1024], K=320 (padded)
        wmma_gemm_kernel<2><<<gr((long)gTiles * 32), 256, 0, stream>>>(
            xcat, KCAT, wcatT, G4H, 10, G4H / 64, gTiles, b, gates, G4H);
        update_kernel<<<gr((long)N_NODES * HDIM), 256, 0, stream>>>(gates, diff, c, h, hbf, xcat);
    }

    out_kernel<<<N_NODES, 256, 0, stream>>>(h, wd, bd, (float*)d_out);
}
